// RegionProposalNetwork_15504831939157
// MI455X (gfx1250) — matrix operand order repaired and checked by reading, not compiled, not run on hardware
//
#include <hip/hip_runtime.h>
#include <hip/hip_bf16.h>

// ---------------- types ----------------
typedef __attribute__((ext_vector_type(16))) __bf16 v16bf;
typedef __attribute__((ext_vector_type(8)))  float  v8f;

#define NEGINF (-__builtin_inff())

typedef __attribute__((address_space(1))) void* as1_vp;
typedef __attribute__((address_space(3))) void* as3_vp;

#if __has_builtin(__builtin_amdgcn_global_load_async_to_lds_b128) && \
    __has_builtin(__builtin_amdgcn_s_wait_asynccnt)
#define HAVE_ASYNC_LDS 1
typedef int vi4 __attribute__((vector_size(16)));
typedef __attribute__((address_space(1))) vi4* as1_v4p;
typedef __attribute__((address_space(3))) vi4* as3_v4p;

__device__ __forceinline__ void async_ld16(const unsigned short* g, unsigned short* l) {
  void* gnc = (void*)g;                       // drop const
  as1_vp g1 = (as1_vp)gnc;                    // generic -> global AS
  as3_vp l3 = (as3_vp)(void*)l;               // generic -> LDS AS
  __builtin_amdgcn_global_load_async_to_lds_b128((as1_v4p)g1, (as3_v4p)l3, 0, 0);
}
#else
#define HAVE_ASYNC_LDS 0
#endif

// TDM: ROCm 7.2 exposes a 5-arg builtin; the toolchain shipping amd_gfx1250_TDM.h uses 6 args
// (extra zeroed int32x8 before cpol).
#if HAVE_ASYNC_LDS && __has_builtin(__builtin_amdgcn_tensor_load_to_lds) && \
    __has_builtin(__builtin_amdgcn_s_wait_tensorcnt)
#define HAVE_TDM 1
typedef unsigned int u32x4 __attribute__((ext_vector_type(4)));
typedef int i32x8 __attribute__((ext_vector_type(8)));
typedef int i32x4 __attribute__((ext_vector_type(4)));

// 2D tile load: rows x 32 bf16 elements (64B/row), row stride in elements; LDS dst contiguous.
__device__ __forceinline__ void tdm_load_2d(const unsigned short* gsrc, unsigned short* ldst,
                                            int rows, int row_stride_elems) {
  const unsigned long long ga = (unsigned long long)gsrc;
  const unsigned lds = (unsigned)(unsigned long long)(as3_vp)(void*)ldst;
  u32x4 g0;
  g0[0] = 1u;                                          // count=1 valid descriptor
  g0[1] = lds;                                         // lds_addr (bytes)
  g0[2] = (unsigned)(ga & 0xFFFFFFFFu);                // global_addr[31:0]
  g0[3] = (unsigned)((ga >> 32) & 0x01FFFFFFu) | (2u << 30);  // global_addr[56:32] | type=2
  i32x8 g1;
  g1[0] = 0x00010000;                                  // data_size=1 (2 bytes/elem)
  g1[1] = 0;                                           // tensor_dim0[15:0]=0 (dim0=1<<30)
  g1[2] = 0x00004000;                                  // tensor_dim0[31:16]; dim1 lo16=0
  g1[3] = 0x00004000 | (32 << 16);                     // dim1 hi16 (1<<30); tile_dim0=32
  g1[4] = rows;                                        // tile_dim1=rows; tile_dim2=0
  g1[5] = row_stride_elems;                            // tensor_dim0_stride[31:0]
  g1[6] = 0;
  g1[7] = 0;
  i32x4 gz = {0, 0, 0, 0};
#if __has_include(<hip/amd_detail/amd_gfx1250_TDM.h>)
  i32x8 gz8 = {0, 0, 0, 0, 0, 0, 0, 0};
  __builtin_amdgcn_tensor_load_to_lds(g0, g1, gz, gz, gz8, 0);
#else
  __builtin_amdgcn_tensor_load_to_lds(g0, g1, gz, gz, 0);
#endif
}
#else
#define HAVE_TDM 0
#endif

__device__ __forceinline__ unsigned short f2bf(float f) {
  union { float f; unsigned u; } v; v.f = f;
  unsigned r = v.u + 0x7FFFu + ((v.u >> 16) & 1u);   // round-to-nearest-even
  return (unsigned short)(r >> 16);
}

// branchless pointer select (avoids exec-mask branches in the pipelined loop)
__device__ __forceinline__ const unsigned short* sel_ptr(bool c, const unsigned short* a,
                                                         const unsigned short* b) {
  unsigned long long m = c ? ~0ull : 0ull;
  return (const unsigned short*)(((unsigned long long)a & m) |
                                 ((unsigned long long)b & ~m));
}

// A fragment: two 16B halves from natural [row][32] tile (contiguous k = 16*lhalf + s, same order as B)
union FragA { v16bf v; uint4 q[2]; };

// ---------------- problem constants ----------------
static constexpr int N_IMG = 8, C_IN = 512, H = 64, W = 64;
static constexpr int A_ANCH = 9;
static constexpr int HWA = H * W * A_ANCH;          // 36864
static constexpr int PIX_TOT = N_IMG * H * W;       // 32768
static constexpr int POST_NMS = 300;

// d_out layout (float elements), reference return order:
static constexpr size_t OUT_DELTAS = 0;                                  // [N*HWA*4]
static constexpr size_t OUT_SCORES = (size_t)N_IMG * HWA * 4;            // [N*HWA*2]
static constexpr size_t OUT_ROIS   = OUT_SCORES + (size_t)N_IMG * HWA * 2;
static constexpr size_t OUT_RIDX   = OUT_ROIS + (size_t)N_IMG * POST_NMS * 4;
static constexpr size_t OUT_ANCH   = OUT_RIDX + (size_t)N_IMG * POST_NMS;

// workspace layout (bytes)
static constexpr size_t WS_XB   = 0;                        // x bf16 NHWC, 32MB
static constexpr size_t WS_WP   = 33554432;                 // conv1 w bf16 [9][512][512]
static constexpr size_t WS_WP2  = WS_WP + 4718592;          // fused 1x1 w bf16 [64][512]
static constexpr size_t WS_ZPAD = WS_WP2 + 65536;           // 4KB zeros (border reads)
static constexpr size_t WS_HID  = WS_ZPAD + 4096;           // hidden bf16 NHWC, 32MB
// boxes/scores alias the x region (x dead after conv3x3):
static constexpr size_t WS_BOX = 0;                         // [8*36864*4] f32
static constexpr size_t WS_SC  = 4718592;                   // [8*36864]   f32

// ---------------- pack kernels (fp32 -> bf16, WMMA-friendly layouts) ----------------
__global__ void pack_x_kernel(const float* __restrict__ x, unsigned short* __restrict__ xb) {
  size_t i = (size_t)blockIdx.x * 256 + threadIdx.x;   // over N*C*H*W, input-indexed
  int xx = (int)(i & 63); size_t r = i >> 6;
  int yy = (int)(r & 63); r >>= 6;
  int c  = (int)(r & 511); r >>= 9;
  int n  = (int)r;
  xb[(((size_t)(n * 64 + yy) * 64 + xx) * 512) + c] = f2bf(x[i]);
}

__global__ void pack_w_kernel(const float* __restrict__ w, unsigned short* __restrict__ wp) {
  size_t i = (size_t)blockIdx.x * 256 + threadIdx.x;   // over 512*512*9, input [cout][ci][3][3]
  int tap = (int)(i % 9); size_t q = i / 9;
  int ci  = (int)(q & 511);
  int cout = (int)(q >> 9);
  wp[((size_t)tap * 512 + cout) * 512 + ci] = f2bf(w[i]);
}

__global__ void pack_w2_kernel(const float* __restrict__ w_pred,
                               const float* __restrict__ w_score,
                               unsigned short* __restrict__ wp2,
                               unsigned short* __restrict__ zpad) {
  int i = blockIdx.x * 256 + threadIdx.x;              // over 64*512
  if (blockIdx.x == 0) ((uint4*)zpad)[threadIdx.x] = make_uint4(0u, 0u, 0u, 0u);
  int co = i >> 9, ci = i & 511;
  float v = 0.0f;
  if (co < 36)      v = w_pred[co * 512 + ci];
  else if (co < 54) v = w_score[(co - 36) * 512 + ci];
  wp2[i] = f2bf(v);
}

// ---------------- 3x3 conv as implicit GEMM, bf16 WMMA ----------------
// Workgroup = (n, y-row, cout-block of 128). C-tile: 64 pixels (x) x 128 couts.
// 144 K-stages of 32; double-buffered LDS; A via async global->LDS, B via TDM (wave 0).
__global__ __launch_bounds__(256) void conv3x3_wmma(
    const unsigned short* __restrict__ xb,    // [8][64][64][512] bf16 NHWC
    const unsigned short* __restrict__ wp,    // [9][512cout][512ci] bf16
    const unsigned short* __restrict__ zpad,  // zeroed pad for out-of-image reads
    const float* __restrict__ bias,           // [512]
    unsigned short* __restrict__ hid)         // [8][64][64][512] bf16 NHWC
{
  __shared__ alignas(64) unsigned short As[2][64][32];    // [buf][pixel][k]
  __shared__ alignas(64) unsigned short Bs[2][128][32];   // [buf][cout][k]
  __shared__ alignas(64) unsigned short Cs[64][128];      // epilogue staging

  const int bid = blockIdx.x;
  const int cb  = bid & 3;             // cout block (128 couts)
  const int y   = (bid >> 2) & 63;
  const int n   = bid >> 8;

  const int t     = threadIdx.x;
  const int lane  = t & 31;
  const int wv    = t >> 5;            // wave 0..7
  const int mi    = wv & 3;            // M (pixel) tile
  const int g     = wv >> 2;           // N group: tiles g*4 .. g*4+3
  const int lhalf = lane >> 4;
  const int lrow  = lane & 15;

  const int p  = t >> 2;               // A staging: pixel
  const int ch = t & 3;                //            8-channel chunk
  const unsigned short* bptr0 = wp + ((size_t)(cb * 128 + (t >> 2)) * 512) + (t & 3) * 8;

  v8f acc[4] = {};

  // stage s: tap = s>>4, ci0 = (s&15)<<5
  auto stage = [&](int s, int b) {
    const int tap = s >> 4;
    const int ci0 = (s & 15) << 5;
    const int ky = tap / 3, kx = tap - 3 * ky;
    const int yy = y + ky - 1, xx = p + kx - 1;
    const bool aok = ((unsigned)yy < 64u) & ((unsigned)xx < 64u);
    const unsigned short* xsrc =
        xb + ((size_t)((n * 64 + yy) * 64 + xx) * 512) + ch * 8 + ci0;
    const unsigned short* asrc = sel_ptr(aok, xsrc, zpad);   // branchless zero fill
#if HAVE_TDM
    async_ld16(asrc, &As[b][p][ch * 8]);
    if (wv == 0)
      tdm_load_2d(wp + ((size_t)tap * 512 + cb * 128) * 512 + ci0, &Bs[b][0][0], 128, 512);
#elif HAVE_ASYNC_LDS
    const unsigned short* bsrc = bptr0 + (size_t)tap * (512 * 512) + ci0;
    async_ld16(asrc, &As[b][p][ch * 8]);
    async_ld16(bsrc,            &Bs[b][t >> 2][(t & 3) * 8]);
    async_ld16(bsrc + 64 * 512, &Bs[b][64 + (t >> 2)][(t & 3) * 8]);
#else
    const unsigned short* bsrc = bptr0 + (size_t)tap * (512 * 512) + ci0;
    *(uint4*)&As[b][p][ch * 8]                  = *(const uint4*)asrc;
    *(uint4*)&Bs[b][t >> 2][(t & 3) * 8]        = *(const uint4*)bsrc;
    *(uint4*)&Bs[b][64 + (t >> 2)][(t & 3) * 8] = *(const uint4*)(bsrc + 64 * 512);
#endif
  };

  stage(0, 0);                         // prologue
#pragma unroll 2
  for (int s = 0; s < 144; ++s) {
    const int b = s & 1;
    if (s < 143) {
      stage(s + 1, b ^ 1);             // next stage's transfers stay in flight
#if HAVE_TDM
      __builtin_amdgcn_s_wait_asynccnt(1);
      if (wv == 0) __builtin_amdgcn_s_wait_tensorcnt(1);
#elif HAVE_ASYNC_LDS
      __builtin_amdgcn_s_wait_asynccnt(3);
#endif
    } else {
#if HAVE_TDM
      __builtin_amdgcn_s_wait_asynccnt(0);
      if (wv == 0) __builtin_amdgcn_s_wait_tensorcnt(0);
#elif HAVE_ASYNC_LDS
      __builtin_amdgcn_s_wait_asynccnt(0);
#endif
    }
    __syncthreads();
    FragA fa;
    fa.q[0] = *(const uint4*)&As[b][mi * 16 + lrow][lhalf * 16];       // K 0..7 / 16..23
    fa.q[1] = *(const uint4*)&As[b][mi * 16 + lrow][lhalf * 16 + 8];   // K 8..15 / 24..31
#pragma unroll
    for (int j = 0; j < 4; ++j) {
      v16bf bf_ = *(const v16bf*)&Bs[b][(g * 4 + j) * 16 + lrow][lhalf * 16];
      acc[j] = __builtin_amdgcn_wmma_f32_16x16x32_bf16(
          false, fa.v, false, bf_, (short)0, acc[j], false, false);
    }
    __syncthreads();
  }

  // epilogue: bias + ReLU -> LDS -> coalesced b128 stores (256B per pixel row)
#pragma unroll
  for (int j = 0; j < 4; ++j) {
    const int coutl = (g * 4 + j) * 16 + lrow;
    const float bv = bias[cb * 128 + coutl];
#pragma unroll
    for (int v = 0; v < 8; ++v) {
      const int px = mi * 16 + v + lhalf * 8;
      Cs[px][coutl] = f2bf(fmaxf(acc[j][v] + bv, 0.0f));
    }
  }
  __syncthreads();
  const size_t rowbase = (size_t)(n * 64 + y) * 64 * 512 + cb * 128;
#pragma unroll
  for (int i = 0; i < 4; ++i) {
    const int idx = t + (i << 8);      // 1024 x 16B chunks
    const int pr  = idx >> 4;          // pixel 0..63
    const int off = (idx & 15) * 8;    // ushort offset in 128-cout row
    *(uint4*)(hid + rowbase + (size_t)pr * 512 + off) = *(const uint4*)&Cs[pr][off];
  }
}

// ---------------- fused 1x1 convs (pred 36ch + score 18ch) as one GEMM ----------------
// C-tile: 64 pixels x 64 cols (0..35 pred, 36..53 score, 54..63 pad). K=512, 16 stages.
__global__ __launch_bounds__(256) void conv1x1_wmma(
    const unsigned short* __restrict__ hid,   // [32768][512] bf16
    const unsigned short* __restrict__ wp2,   // [64][512] bf16
    const float* __restrict__ b_score,
    const float* __restrict__ b_pred,
    float* __restrict__ out)
{
  __shared__ alignas(64) unsigned short As[2][64][32];
  __shared__ alignas(64) unsigned short Bs[2][64][32];

  const int p0 = blockIdx.x * 64;
  const int t = threadIdx.x, lane = t & 31, wv = t >> 5;
  const int mi = wv & 3, g = wv >> 2;         // N tiles: g*2, g*2+1
  const int lhalf = lane >> 4, lrow = lane & 15;
  const int p = t >> 2, ch = t & 3;

  v8f acc[2] = {};

  auto stage = [&](int s, int b) {
    const int k0 = s << 5;
#if HAVE_TDM
    if (wv == 0) {
      tdm_load_2d(hid + (size_t)p0 * 512 + k0, &As[b][0][0], 64, 512);
      tdm_load_2d(wp2 + k0,                    &Bs[b][0][0], 64, 512);
    }
#elif HAVE_ASYNC_LDS
    async_ld16(hid + (size_t)(p0 + p) * 512 + k0 + ch * 8, &As[b][p][ch * 8]);
    async_ld16(wp2 + (size_t)(t >> 2) * 512 + k0 + (t & 3) * 8, &Bs[b][t >> 2][(t & 3) * 8]);
#else
    *(uint4*)&As[b][p][ch * 8] = *(const uint4*)(hid + (size_t)(p0 + p) * 512 + k0 + ch * 8);
    *(uint4*)&Bs[b][t >> 2][(t & 3) * 8] =
        *(const uint4*)(wp2 + (size_t)(t >> 2) * 512 + k0 + (t & 3) * 8);
#endif
  };

  stage(0, 0);
#pragma unroll 2
  for (int s = 0; s < 16; ++s) {
    const int b = s & 1;
    if (s < 15) {
      stage(s + 1, b ^ 1);
#if HAVE_TDM
      if (wv == 0) __builtin_amdgcn_s_wait_tensorcnt(2);
#elif HAVE_ASYNC_LDS
      __builtin_amdgcn_s_wait_asynccnt(2);
#endif
    } else {
#if HAVE_TDM
      if (wv == 0) __builtin_amdgcn_s_wait_tensorcnt(0);
#elif HAVE_ASYNC_LDS
      __builtin_amdgcn_s_wait_asynccnt(0);
#endif
    }
    __syncthreads();
    FragA fa;
    fa.q[0] = *(const uint4*)&As[b][mi * 16 + lrow][lhalf * 16];
    fa.q[1] = *(const uint4*)&As[b][mi * 16 + lrow][lhalf * 16 + 8];
#pragma unroll
    for (int j = 0; j < 2; ++j) {
      v16bf bf_ = *(const v16bf*)&Bs[b][(g * 2 + j) * 16 + lrow][lhalf * 16];
      acc[j] = __builtin_amdgcn_wmma_f32_16x16x32_bf16(
          false, fa.v, false, bf_, (short)0, acc[j], false, false);
    }
    __syncthreads();
  }

#pragma unroll
  for (int j = 0; j < 2; ++j) {
    const int cp = (g * 2 + j) * 16 + lrow;      // fused channel 0..63
    const float bv = (cp < 36) ? b_pred[cp] : ((cp < 54) ? b_score[cp - 36] : 0.0f);
#pragma unroll
    for (int v = 0; v < 8; ++v) {
      const int px = p0 + mi * 16 + v + lhalf * 8;
      const float val = acc[j][v] + bv;
      if (cp < 36)       out[OUT_DELTAS + (size_t)px * 36 + cp] = val;
      else if (cp < 54)  out[OUT_SCORES + (size_t)px * 18 + (cp - 36)] = val;
    }
  }
}

// ---------------- anchors + softmax-fg + decode + clip + min-size ----------------
__global__ void decode_boxes(float* __restrict__ out,
                             float* __restrict__ boxes,
                             float* __restrict__ sc,
                             const int* __restrict__ img_h_p,
                             const int* __restrict__ img_w_p)
{
  const int idx = blockIdx.x * 256 + threadIdx.x;
  if (idx >= N_IMG * HWA) return;
  const int n = idx / HWA, i = idx - n * HWA;
  const int a = i % 9, pix = i / 9;
  const int x = pix & 63, yv = pix >> 6;
  const int ri = a / 3, si = a - 3 * ri;
  const float ratio = (ri == 0) ? 0.5f : (ri == 1 ? 1.0f : 2.0f);
  const float scale = (si == 0) ? 8.0f : (si == 1 ? 16.0f : 32.0f);
  const float hs  = 16.0f * scale * __builtin_sqrtf(ratio);
  const float ws_ = 16.0f * scale * __builtin_sqrtf(1.0f / ratio);
  const float acy = yv * 16.0f + 8.0f, acx = x * 16.0f + 8.0f;

  if (n == 0) {
    out[OUT_ANCH + (size_t)i * 4 + 0] = acy - 0.5f * hs;
    out[OUT_ANCH + (size_t)i * 4 + 1] = acx - 0.5f * ws_;
    out[OUT_ANCH + (size_t)i * 4 + 2] = acy + 0.5f * hs;
    out[OUT_ANCH + (size_t)i * 4 + 3] = acx + 0.5f * ws_;
  }

  const size_t db = OUT_DELTAS + (size_t)idx * 4;
  const float dy = out[db], dx = out[db + 1], dh = out[db + 2], dw = out[db + 3];
  const float cy = dy * hs + acy, cx = dx * ws_ + acx;
  const float hh = expf(dh) * hs, ww = expf(dw) * ws_;
  const float imH = (float)img_h_p[0], imW = (float)img_w_p[0];
  const float y1 = fminf(fmaxf(cy - 0.5f * hh, 0.0f), imH);
  const float y2 = fminf(fmaxf(cy + 0.5f * hh, 0.0f), imH);
  const float x1 = fminf(fmaxf(cx - 0.5f * ww, 0.0f), imW);
  const float x2 = fminf(fmaxf(cx + 0.5f * ww, 0.0f), imW);
  const bool valid = ((y2 - y1) >= 16.0f) & ((x2 - x1) >= 16.0f);

  const float s0 = out[OUT_SCORES + (size_t)idx * 2];
  const float s1 = out[OUT_SCORES + (size_t)idx * 2 + 1];
  const float fg = 1.0f / (1.0f + expf(s0 - s1));

  sc[idx] = valid ? fg : NEGINF;
  boxes[(size_t)idx * 4 + 0] = y1;
  boxes[(size_t)idx * 4 + 1] = x1;
  boxes[(size_t)idx * 4 + 2] = y2;
  boxes[(size_t)idx * 4 + 3] = x2;
}

// ---------------- greedy NMS: one workgroup per image, 300 sequential rounds ----------------
__global__ __launch_bounds__(1024) void nms_kernel(const float* __restrict__ boxes,
                                                   float* __restrict__ sc,
                                                   float* __restrict__ out)
{
  const int n = blockIdx.x;
  float* S = sc + (size_t)n * HWA;
  const float* B = boxes + (size_t)n * HWA * 4;

  __shared__ float sv[1024];
  __shared__ int   si[1024];
  __shared__ float cbox[5];
  const int t = threadIdx.x;

  for (int it = 0; it < POST_NMS; ++it) {
    float bestv = NEGINF; int besti = -1;
    for (int k = t; k < HWA; k += 1024) {
      const float v = S[k];
      if (v > bestv) { bestv = v; besti = k; }
    }
    sv[t] = bestv; si[t] = besti;
    __syncthreads();
    for (int s = 512; s > 0; s >>= 1) {
      if (t < s) {
        const bool take = (sv[t + s] > sv[t]) ||
                          (sv[t + s] == sv[t] && si[t + s] != -1 &&
                           (si[t] == -1 || si[t + s] < si[t]));
        if (take) { sv[t] = sv[t + s]; si[t] = si[t + s]; }
      }
      __syncthreads();
    }
    if (t == 0) {
      const int j = si[0];
      const bool ok = (j >= 0);
      float y1 = 0.f, x1 = 0.f, y2 = 0.f, x2 = 0.f;
      if (ok) {
        y1 = B[(size_t)j * 4]; x1 = B[(size_t)j * 4 + 1];
        y2 = B[(size_t)j * 4 + 2]; x2 = B[(size_t)j * 4 + 3];
        S[j] = NEGINF;
      }
      cbox[0] = y1; cbox[1] = x1; cbox[2] = y2; cbox[3] = x2; cbox[4] = ok ? 1.0f : 0.0f;
      float* ro = out + OUT_ROIS + ((size_t)n * POST_NMS + it) * 4;
      ro[0] = y1; ro[1] = x1; ro[2] = y2; ro[3] = x2;
      out[OUT_RIDX + (size_t)n * POST_NMS + it] = (float)n;
    }
    __syncthreads();
    if (cbox[4] > 0.0f) {
      const float jy1 = cbox[0], jx1 = cbox[1], jy2 = cbox[2], jx2 = cbox[3];
      const float ja = (jy2 - jy1) * (jx2 - jx1);
      for (int k = t; k < HWA; k += 1024) {
        if (S[k] > NEGINF) {
          const float by1 = B[(size_t)k * 4],     bx1 = B[(size_t)k * 4 + 1];
          const float by2 = B[(size_t)k * 4 + 2], bx2 = B[(size_t)k * 4 + 3];
          const float yy1 = fmaxf(jy1, by1), xx1 = fmaxf(jx1, bx1);
          const float yy2 = fminf(jy2, by2), xx2 = fminf(jx2, bx2);
          const float inter = fmaxf(yy2 - yy1, 0.0f) * fmaxf(xx2 - xx1, 0.0f);
          const float area = (by2 - by1) * (bx2 - bx1);
          if (inter / (ja + area - inter) > 0.7f) S[k] = NEGINF;
        }
      }
    }
    __syncthreads();
  }
}

// ---------------- host launch ----------------
extern "C" void kernel_launch(void* const* d_in, const int* in_sizes, int n_in,
                              void* d_out, int out_size, void* d_ws, size_t ws_size,
                              hipStream_t stream) {
  const float* x       = (const float*)d_in[0];
  const float* w_conv1 = (const float*)d_in[1];
  const float* b_conv1 = (const float*)d_in[2];
  const float* w_score = (const float*)d_in[3];
  const float* b_score = (const float*)d_in[4];
  const float* w_pred  = (const float*)d_in[5];
  const float* b_pred  = (const float*)d_in[6];
  const int*   img_h   = (const int*)d_in[7];
  const int*   img_w   = (const int*)d_in[8];
  float* out = (float*)d_out;
  char*  ws  = (char*)d_ws;

  unsigned short* xb   = (unsigned short*)(ws + WS_XB);
  unsigned short* wp   = (unsigned short*)(ws + WS_WP);
  unsigned short* wp2  = (unsigned short*)(ws + WS_WP2);
  unsigned short* zpad = (unsigned short*)(ws + WS_ZPAD);
  unsigned short* hid  = (unsigned short*)(ws + WS_HID);
  float* boxes = (float*)(ws + WS_BOX);   // aliases xb (x dead after conv3x3)
  float* scb   = (float*)(ws + WS_SC);

  pack_x_kernel<<<65536, 256, 0, stream>>>(x, xb);
  pack_w_kernel<<<9216, 256, 0, stream>>>(w_conv1, wp);
  pack_w2_kernel<<<128, 256, 0, stream>>>(w_pred, w_score, wp2, zpad);

  conv3x3_wmma<<<N_IMG * 64 * 4, 256, 0, stream>>>(xb, wp, zpad, b_conv1, hid);
  conv1x1_wmma<<<PIX_TOT / 64, 256, 0, stream>>>(hid, wp2, b_score, b_pred, out);

  decode_boxes<<<(N_IMG * HWA + 255) / 256, 256, 0, stream>>>(out, boxes, scb, img_h, img_w);
  nms_kernel<<<N_IMG, 1024, 0, stream>>>(boxes, scb, out);
}
